// ImprovedNeighborEmbedding_77489799954761
// MI455X (gfx1250) — compile-verified, matrix operand order checked by reading, not run on hardware
//
#include <hip/hip_runtime.h>
#include <hip/hip_bf16.h>
#include <math.h>

#define D 128
#define TILES_PER_BLOCK 8

typedef __attribute__((ext_vector_type(16))) __bf16 v16bf;
typedef __attribute__((ext_vector_type(8)))  float  v8f;

__device__ __forceinline__ unsigned short f2bf(float f) {
    unsigned u = __float_as_uint(f);
    u += 0x7FFFu + ((u >> 16) & 1u);          // round-to-nearest-even
    return (unsigned short)(u >> 16);
}
// order-preserving encode for float atomicMax on uint
__device__ __forceinline__ unsigned enc_f(float f) {
    unsigned u = __float_as_uint(f);
    return (u & 0x80000000u) ? ~u : (u | 0x80000000u);
}
__device__ __forceinline__ float dec_f(unsigned e) {
    return (e & 0x80000000u) ? __uint_as_float(e & 0x7FFFFFFFu)
                             : __uint_as_float(~e);
}

// ---------------------------------------------------------------- zero init
__global__ void k_zero(unsigned* p, long long n) {
    long long i = (long long)blockIdx.x * blockDim.x + threadIdx.x;
    long long stride = (long long)gridDim.x * blockDim.x;
    for (; i < n; i += stride) p[i] = 0u;
}

// ------------------------------------------- repack weights to WMMA-B bf16
// Layout: Wpk[(((w*8+nt)*4+kf)*32+lane)*16 + j], element j of lane's v16bf.
// B[k][n]: n = nt*16 + (lane&15), kh = lane>>4,
// k = kf*32 + (j<8 ? kh*8 + j : 16 + kh*8 + (j-8))
__global__ void k_repack(const float* W0, const float* W1, const float* W2,
                         const float* W3, const float* W4, unsigned short* Wpk) {
    int t = blockIdx.x * blockDim.x + threadIdx.x;
    if (t >= 5 * 8 * 4 * 32 * 16) return;
    int j    = t & 15;
    int lane = (t >> 4) & 31;
    int kf   = (t >> 9) & 3;
    int nt   = (t >> 11) & 7;
    int w    = t >> 14;
    const float* W = (w == 0) ? W0 : (w == 1) ? W1 : (w == 2) ? W2 : (w == 3) ? W3 : W4;
    int n  = nt * 16 + (lane & 15);
    int kh = lane >> 4;
    int k  = kf * 32 + ((j < 8) ? (kh * 8 + j) : (16 + kh * 8 + (j - 8)));
    Wpk[t] = f2bf(W[k * D + n]);
}

// ---------------------------------------------- fused 5-way WMMA GEMM (bf16)
// Each wave: one 16-row tile of embedding; computes h,q,k,v,r tiles.
__global__ __launch_bounds__(256) void k_gemm(const float* __restrict__ emb,
                                              const unsigned short* __restrict__ Wpk,
                                              float* h, float* q, float* kk,
                                              float* v, float* r, int nTiles) {
    __shared__ unsigned short Abf[TILES_PER_BLOCK][16 * 130]; // stride 130 kills bank conflicts
    int wave = threadIdx.x >> 5;
    int lane = threadIdx.x & 31;
    int tIdx = blockIdx.x * TILES_PER_BLOCK + wave;
    bool active = tIdx < nTiles;
    int mbase = tIdx * 16;

    if (active) {
        // cooperative f32->bf16 stage of the 16x128 A tile
        int row = lane >> 1, cb = (lane & 1) * 64;
        const float* src = emb + (long long)(mbase + row) * D + cb;
        unsigned short* dst = &Abf[wave][row * 130 + cb];
        #pragma unroll
        for (int t = 0; t < 16; ++t) {
            float4 f = *(const float4*)(src + t * 4);
            unsigned lo = (unsigned)f2bf(f.x) | ((unsigned)f2bf(f.y) << 16);
            unsigned hi = (unsigned)f2bf(f.z) | ((unsigned)f2bf(f.w) << 16);
            *(unsigned*)(dst + t * 4)     = lo;
            *(unsigned*)(dst + t * 4 + 2) = hi;
        }
    }
    __syncthreads();
    if (!active) return;

    int m = lane & 15, kh = lane >> 4;
    union { v16bf v; unsigned u[8]; } af[4];
    const unsigned short* arow = &Abf[wave][m * 130];
    #pragma unroll
    for (int kf = 0; kf < 4; ++kf)
        #pragma unroll
        for (int vg = 0; vg < 8; ++vg) {
            int k0 = kf * 32 + ((vg >> 2) << 4) + kh * 8 + ((vg & 3) << 1);
            af[kf].u[vg] = *(const unsigned*)(arow + k0);
        }

    float* outs[5] = {h, q, kk, v, r};
    for (int w = 0; w < 5; ++w) {
        float* o = outs[w];
        for (int nt = 0; nt < 8; ++nt) {
            v8f acc = {0.f, 0.f, 0.f, 0.f, 0.f, 0.f, 0.f, 0.f};
            const unsigned short* bp = Wpk + (((size_t)(w * 8 + nt) * 4) * 32 + lane) * 16;
            #pragma unroll
            for (int kf = 0; kf < 4; ++kf) {
                v16bf b = *(const v16bf*)(bp + (size_t)kf * 32 * 16);
                acc = __builtin_amdgcn_wmma_f32_16x16x32_bf16(
                          false, af[kf].v, false, b, (short)0, acc, false, false);
            }
            long long base = (long long)mbase * D + nt * 16 + m;
            #pragma unroll
            for (int i = 0; i < 8; ++i)
                o[base + (long long)(i + 8 * kh) * D] = acc[i];
        }
    }
}

// ------------------------------------------- per-node attention coefficients
__global__ __launch_bounds__(256) void k_attcoef(const float* __restrict__ h,
                                                 const float* __restrict__ a_src,
                                                 const float* __restrict__ a_dst,
                                                 float* a_s, float* a_d, int N_) {
    int node = (int)(((long long)blockIdx.x * blockDim.x + threadIdx.x) >> 5);
    int lane = threadIdx.x & 31;
    if (node >= N_) return;
    float4 hv  = *(const float4*)(h + (long long)node * D + lane * 4);
    float4 as4 = *(const float4*)(a_src + lane * 4);
    float4 ad4 = *(const float4*)(a_dst + lane * 4);
    float s = hv.x * as4.x + hv.y * as4.y + hv.z * as4.z + hv.w * as4.w;
    float d = hv.x * ad4.x + hv.y * ad4.y + hv.z * ad4.z + hv.w * ad4.w;
    for (int off = 16; off; off >>= 1) {
        s += __shfl_down(s, off);
        d += __shfl_down(d, off);
    }
    if (lane == 0) { a_s[node] = s; a_d[node] = d; }
}

// ------------------------------------ per-edge logits + segment max (atomic)
__global__ __launch_bounds__(256) void k_logits(const int* __restrict__ edge, long long E_,
                                                const float* __restrict__ q,
                                                const float* __restrict__ k,
                                                const float* __restrict__ a_s,
                                                const float* __restrict__ a_d,
                                                float* lg, float* lt,
                                                unsigned* mg, unsigned* mt) {
    long long e = ((long long)blockIdx.x * blockDim.x + threadIdx.x) >> 5;
    int lane = threadIdx.x & 31;
    if (e >= E_) return;
    int s  = edge[e];
    int dd = edge[E_ + e];
    float4 qv = *(const float4*)(q + (long long)dd * D + lane * 4);
    float4 kv = *(const float4*)(k + (long long)s  * D + lane * 4);
    float p = qv.x * kv.x + qv.y * kv.y + qv.z * kv.z + qv.w * kv.w;
    for (int off = 16; off; off >>= 1) p += __shfl_down(p, off);
    if (lane == 0) {
        float l1 = a_s[s] + a_d[dd];
        l1 = (l1 >= 0.f) ? l1 : 0.2f * l1;           // LeakyReLU(0.2)
        lg[e] = l1;
        atomicMax(&mg[dd], enc_f(l1));
        float l2 = p * 0.08838834764831845f;          // 1/sqrt(128)
        lt[e] = l2;
        atomicMax(&mt[dd], enc_f(l2));
    }
}

// -------------------------------------------------- exp + segment sum (Z)
__global__ void k_expsum(const int* __restrict__ edge, long long E_,
                         float* lg, float* lt,
                         const unsigned* __restrict__ mg, const unsigned* __restrict__ mt,
                         float* zg, float* zt) {
    long long e = (long long)blockIdx.x * blockDim.x + threadIdx.x;
    if (e >= E_) return;
    int dd = edge[E_ + e];
    float eg = expf(lg[e] - dec_f(mg[dd]));
    lg[e] = eg;
    atomicAdd(&zg[dd], eg);
    float et = expf(lt[e] - dec_f(mt[dd]));
    lt[e] = et;
    atomicAdd(&zt[dd], et);
}

// --------------------------------------- weighted scatter (unnormalized sums)
__global__ __launch_bounds__(256) void k_scatter(const int* __restrict__ edge, long long E_,
                                                 const float* __restrict__ lg,
                                                 const float* __restrict__ lt,
                                                 const float* __restrict__ h,
                                                 const float* __restrict__ v,
                                                 float* accg, float* acct) {
    long long e = ((long long)blockIdx.x * blockDim.x + threadIdx.x) >> 5;
    int lane = threadIdx.x & 31;
    if (e >= E_) return;
    int s  = edge[e];
    int dd = edge[E_ + e];
    float eg = lg[e];
    float et = lt[e];
    int c = lane * 4;
    float4 hv = *(const float4*)(h + (long long)s * D + c);
    float4 vv = *(const float4*)(v + (long long)s * D + c);
    float* pg = accg + (long long)dd * D + c;
    float* pt = acct + (long long)dd * D + c;
    atomicAdd(&pg[0], eg * hv.x);
    atomicAdd(&pg[1], eg * hv.y);
    atomicAdd(&pg[2], eg * hv.z);
    atomicAdd(&pg[3], eg * hv.w);
    atomicAdd(&pt[0], et * vv.x);
    atomicAdd(&pt[1], et * vv.y);
    atomicAdd(&pt[2], et * vv.z);
    atomicAdd(&pt[3], et * vv.w);
}

// ----------------------- final: normalize by Z, relu+bias, skip, L2 normalize
__global__ __launch_bounds__(256) void k_final(const float* __restrict__ accg,
                                               const float* __restrict__ acct,
                                               const float* __restrict__ zg,
                                               const float* __restrict__ zt,
                                               const float* __restrict__ bias,
                                               const float* __restrict__ r,
                                               float* out, int N_) {
    int node = (int)(((long long)blockIdx.x * blockDim.x + threadIdx.x) >> 5);
    int lane = threadIdx.x & 31;
    if (node >= N_) return;
    float izg = 1.f / (zg[node] + 1e-16f);
    float izt = 1.f / (zt[node] + 1e-16f);
    int c = lane * 4;
    float4 g  = *(const float4*)(accg + (long long)node * D + c);
    float4 t4 = *(const float4*)(acct + (long long)node * D + c);
    float4 b4 = *(const float4*)(bias + c);
    float4 r4 = *(const float4*)(r + (long long)node * D + c);
    float o0 = fmaxf(g.x * izg + b4.x, 0.f) + (t4.x * izt + r4.x);
    float o1 = fmaxf(g.y * izg + b4.y, 0.f) + (t4.y * izt + r4.y);
    float o2 = fmaxf(g.z * izg + b4.z, 0.f) + (t4.z * izt + r4.z);
    float o3 = fmaxf(g.w * izg + b4.w, 0.f) + (t4.w * izt + r4.w);
    float ss = o0 * o0 + o1 * o1 + o2 * o2 + o3 * o3;
    for (int off = 16; off; off >>= 1) ss += __shfl_down(ss, off);
    ss = __shfl(ss, 0);
    float inv = 1.f / fmaxf(sqrtf(ss), 1e-12f);
    float4 res = make_float4(o0 * inv, o1 * inv, o2 * inv, o3 * inv);
    *(float4*)(out + (long long)node * D + c) = res;
}

// ---------------------------------------------------------------------------
extern "C" void kernel_launch(void* const* d_in, const int* in_sizes, int n_in,
                              void* d_out, int out_size, void* d_ws, size_t ws_size,
                              hipStream_t stream) {
    const int*   edge   = (const int*)d_in[1];
    const float* emb    = (const float*)d_in[2];
    const float* gat_W  = (const float*)d_in[3];
    const float* a_src  = (const float*)d_in[4];
    const float* a_dst  = (const float*)d_in[5];
    const float* g_bias = (const float*)d_in[6];
    const float* Wq     = (const float*)d_in[7];
    const float* Wk     = (const float*)d_in[8];
    const float* Wv     = (const float*)d_in[9];
    const float* Wr     = (const float*)d_in[10];
    float* out = (float*)d_out;

    long long E_ = (long long)in_sizes[1] / 2;
    int       N_ = in_sizes[2] / D;
    size_t ND = (size_t)N_ * D;

    float* ws = (float*)d_ws;
    float* h    = ws;
    float* q    = ws + 1 * ND;
    float* kbuf = ws + 2 * ND;
    float* vbuf = ws + 3 * ND;
    float* rbuf = ws + 4 * ND;
    float* accg = ws + 5 * ND;
    float* acct = ws + 6 * ND;
    float* a_s  = ws + 7 * ND;
    float* a_d  = a_s + N_;
    float* zg   = a_d + N_;
    float* zt   = zg + N_;
    unsigned* mg = (unsigned*)(zt + N_);
    unsigned* mt = mg + N_;
    float* lg   = (float*)(mt + N_);
    float* lt   = lg + E_;
    size_t offW = ((size_t)((lt + E_) - ws) + 7) & ~(size_t)7;   // 32B align
    unsigned short* Wpk = (unsigned short*)(ws + offW);

    // 1) zero: accumulators + Z + segment-max sentinels (0 == encoded -inf)
    k_zero<<<2048, 256, 0, stream>>>((unsigned*)accg, (long long)(2 * ND));
    k_zero<<<256, 256, 0, stream>>>((unsigned*)zg, (long long)(4 * N_));

    // 2) repack 5 weight matrices into bf16 WMMA-B fragments
    k_repack<<<(5 * 8 * 4 * 32 * 16 + 255) / 256, 256, 0, stream>>>(
        gat_W, Wq, Wk, Wv, Wr, Wpk);

    // 3) fused 5-way GEMM: h, q, k, v, r = emb @ {gat_W, Wq, Wk, Wv, Wr}
    int nTiles = (N_ + 15) / 16;
    k_gemm<<<(nTiles + TILES_PER_BLOCK - 1) / TILES_PER_BLOCK, 256, 0, stream>>>(
        emb, Wpk, h, q, kbuf, vbuf, rbuf, nTiles);

    // 4) per-node GAT attention coefficients
    k_attcoef<<<(N_ + 7) / 8, 256, 0, stream>>>(h, a_src, a_dst, a_s, a_d, N_);

    // 5) per-edge logits + segment max
    k_logits<<<(int)((E_ + 7) / 8), 256, 0, stream>>>(edge, E_, q, kbuf, a_s, a_d,
                                                      lg, lt, mg, mt);

    // 6) exp + segment sums
    k_expsum<<<(int)((E_ + 255) / 256), 256, 0, stream>>>(edge, E_, lg, lt, mg, mt, zg, zt);

    // 7) weighted scatter of h[src] / v[src]
    k_scatter<<<(int)((E_ + 7) / 8), 256, 0, stream>>>(edge, E_, lg, lt, h, vbuf, accg, acct);

    // 8) finalize: relu(gat/Z + bias) + (t/Z + r), row L2 normalize
    k_final<<<(N_ + 7) / 8, 256, 0, stream>>>(accg, acct, zg, zt, g_bias, rbuf, out, N_);

    (void)n_in; (void)out_size; (void)ws_size;
}
